// Attention_68418829025870
// MI455X (gfx1250) — matrix operand order repaired and checked by reading, not compiled, not run on hardware
//
#include <hip/hip_runtime.h>
#include <hip/hip_bf16.h>

// ---------------------------------------------------------------------------
// Shapes (fixed by the reference)
// ---------------------------------------------------------------------------
#define NB    2
#define TSEQ  2048
#define TDIM  1024
#define NH    16
#define HDIM  64
#define WIN   512

typedef __attribute__((ext_vector_type(16))) __bf16 v16bf;
typedef __attribute__((ext_vector_type(8)))  float  v8f;
typedef int v4i_ __attribute__((vector_size(16)));

union FragB { v16bf v; unsigned u[8]; };
union FragC { v8f   v; float    f[8]; };
union Stage { uint4 q[2]; unsigned short s[16]; };

#if defined(__AMDGCN__) && __has_builtin(__builtin_amdgcn_global_load_async_to_lds_b128)
#define HAVE_ASYNC 1
#endif

// 16-byte global -> LDS copy. Async path bypasses VGPRs (ASYNCcnt tracked).
static __device__ __forceinline__ void cp16_g2l(unsigned short* l,
                                                const unsigned short* g) {
#ifdef HAVE_ASYNC
  __builtin_amdgcn_global_load_async_to_lds_b128(
      (__attribute__((address_space(1))) v4i_*)(uintptr_t)g,
      (__attribute__((address_space(3))) v4i_*)(unsigned)(uintptr_t)l,
      0, 0);
#else
  *(uint4*)l = *(const uint4*)g;
#endif
}

static __device__ __forceinline__ void wait_async0() {
#ifdef HAVE_ASYNC
#if __has_builtin(__builtin_amdgcn_s_wait_asynccnt)
  __builtin_amdgcn_s_wait_asynccnt(0);
#else
  asm volatile("s_wait_asynccnt 0x0" ::: "memory");
#endif
#endif
}

static __device__ __forceinline__ unsigned short f2bf(float f) {
  union { float f; unsigned u; } x; x.f = f;
  unsigned r = x.u + 0x7FFFu + ((x.u >> 16) & 1u);   // round-to-nearest-even
  return (unsigned short)(r >> 16);
}

// ---------------------------------------------------------------------------
// fp32 -> bf16 bulk convert (n multiple of 4)
// ---------------------------------------------------------------------------
__global__ void cvt_f32_bf16(const float* __restrict__ in,
                             unsigned short* __restrict__ out, int n) {
  int i = (blockIdx.x * blockDim.x + threadIdx.x) * 4;
  if (i + 3 < n) {
    float4 f = *(const float4*)(in + i);
    union { unsigned short s[4]; uint2 u; } o;
    o.s[0] = f2bf(f.x); o.s[1] = f2bf(f.y); o.s[2] = f2bf(f.z); o.s[3] = f2bf(f.w);
    *(uint2*)(out + i) = o.u;
  }
}

// ---------------------------------------------------------------------------
// Tiled WMMA GEMM:  C[M,N] = A[M,K] (bf16) * W[N,K]^T (bf16), fp32 accum.
// Block tile 128(M) x 64(N), 256 threads = 8 waves, wave tile 32x32.
// K-slab 64, double-buffered LDS fed by async global->LDS copies.
// mode 0: write fp32 row-major to out32[M,N]
// mode 1: write bf16 to outbf in [B,H,T,HD] layout (for V)
// ---------------------------------------------------------------------------
__global__ void gemm_bf16(const unsigned short* __restrict__ A,
                          const unsigned short* __restrict__ W,
                          float* __restrict__ out32,
                          unsigned short* __restrict__ outbf,
                          int M, int N, int K, int mode) {
  __shared__ __attribute__((aligned(16))) unsigned short lds_a[2][128 * 64]; // 32 KB
  __shared__ __attribute__((aligned(16))) unsigned short lds_b[2][64 * 64];  // 16 KB

  const int t     = threadIdx.x;
  const int wid   = t >> 5;
  const int lane  = t & 31;
  const int lhalf = lane >> 4;
  const int lmod  = lane & 15;
  const int m0    = blockIdx.x * 128;
  const int n0    = blockIdx.y * 64;

  v8f zf = {};
  v8f acc[2][2];
  acc[0][0] = zf; acc[0][1] = zf; acc[1][0] = zf; acc[1][1] = zf;

  auto load_slab = [&](int buf, int kc) {
    // A: 128x64 bf16 = 16 KB = 1024 x 16B chunks, 4 per thread
#pragma unroll
    for (int r = 0; r < 4; ++r) {
      int idx = t + r * 256;
      int row = idx >> 3;             // 8 chunks per 64-col row
      int cc  = (idx & 7) * 8;
      cp16_g2l(&lds_a[buf][row * 64 + cc], &A[(m0 + row) * K + kc + cc]);
    }
    // B: 64x64 bf16 = 8 KB = 512 chunks, 2 per thread; stored [n][k]
#pragma unroll
    for (int r = 0; r < 2; ++r) {
      int idx = t + r * 256;
      int row = idx >> 3;
      int cc  = (idx & 7) * 8;
      cp16_g2l(&lds_b[buf][row * 64 + cc], &W[(n0 + row) * K + kc + cc]);
    }
  };

  const int nslab = K >> 6;           // K / 64
  load_slab(0, 0);
  wait_async0();
  __syncthreads();

  for (int s = 0; s < nslab; ++s) {
    const int cur = s & 1;
    if (s + 1 < nslab) load_slab(cur ^ 1, (s + 1) * 64);   // stream next slab

    const unsigned* pa = (const unsigned*)&lds_a[cur][0];
    const unsigned* pb = (const unsigned*)&lds_b[cur][0];
#pragma unroll
    for (int kc2 = 0; kc2 < 2; ++kc2) {
      FragB afr[2], bfr[2];
#pragma unroll
      for (int mi = 0; mi < 2; ++mi) {
        int m    = (wid >> 1) * 32 + mi * 16 + lmod;
        int base = (m * 64 + kc2 * 32) >> 1;
        int kb2  = lhalf * 4;                 // (lhalf*8)>>1
#pragma unroll
        for (int j = 0; j < 4; ++j) {
          afr[mi].u[j]     = pa[base + kb2 + j];
          afr[mi].u[4 + j] = pa[base + 8 + kb2 + j];
        }
      }
#pragma unroll
      for (int ni = 0; ni < 2; ++ni) {
        int n    = (wid & 1) * 32 + ni * 16 + lmod;
        int base = (n * 64 + kc2 * 32 + lhalf * 8) >> 1;
#pragma unroll
        for (int j = 0; j < 4; ++j) { bfr[ni].u[j] = pb[base + j]; bfr[ni].u[4 + j] = pb[base + 8 + j]; }
      }
#pragma unroll
      for (int mi = 0; mi < 2; ++mi)
#pragma unroll
        for (int ni = 0; ni < 2; ++ni)
          acc[mi][ni] = __builtin_amdgcn_wmma_f32_16x16x32_bf16(
              false, afr[mi].v, false, bfr[ni].v, (short)0, acc[mi][ni], false, false);
    }
    wait_async0();      // next slab fully landed in LDS
    __syncthreads();    // all waves done reading cur + see next slab
  }

  // epilogue
#pragma unroll
  for (int mi = 0; mi < 2; ++mi) {
#pragma unroll
    for (int ni = 0; ni < 2; ++ni) {
      FragC c; c.v = acc[mi][ni];
      int col = n0 + (wid & 1) * 32 + ni * 16 + lmod;
#pragma unroll
      for (int e = 0; e < 8; ++e) {
        int row = m0 + (wid >> 1) * 32 + mi * 16 + e + lhalf * 8;
        float val = c.f[e];
        if (mode == 1) {
          int b = row / TSEQ, tt = row % TSEQ;
          int h = col >> 6, hd = col & 63;
          outbf[((b * NH + h) * TSEQ + tt) * HDIM + hd] = f2bf(val);
        } else {
          out32[row * N + col] = val;
        }
      }
    }
  }
}

// ---------------------------------------------------------------------------
// RMSNorm over a 1024-wide row + repack bf16 to [B,H,T,HD]
// ---------------------------------------------------------------------------
__global__ void rmsnorm_pack(const float* __restrict__ in,
                             const float* __restrict__ w,
                             unsigned short* __restrict__ out) {
  const int row = blockIdx.x;        // b*T + t
  const int t   = threadIdx.x;       // 256 threads
  __shared__ float red[8];
  __shared__ float scale;

  float4 x = *(const float4*)(in + row * TDIM + t * 4);
  float ss = x.x * x.x + x.y * x.y + x.z * x.z + x.w * x.w;
#pragma unroll
  for (int off = 16; off > 0; off >>= 1) ss += __shfl_down(ss, off, 32);
  if ((t & 31) == 0) red[t >> 5] = ss;
  __syncthreads();
  if (t == 0) {
    float s = 0.f;
#pragma unroll
    for (int i = 0; i < 8; ++i) s += red[i];
    scale = rsqrtf(s / (float)TDIM + 1e-6f);
  }
  __syncthreads();

  const float sc = scale;
  const int b = row / TSEQ, tt = row % TSEQ;
  const float* xp = (const float*)&x;
#pragma unroll
  for (int j = 0; j < 4; ++j) {
    int col = t * 4 + j;
    float val = xp[j] * sc * w[col];
    int h = col >> 6, hd = col & 63;
    out[((b * NH + h) * TSEQ + tt) * HDIM + hd] = f2bf(val);
  }
}

// ---------------------------------------------------------------------------
// Flash attention: one (b, h, 64-query tile) per block. 256 threads = 8 waves.
// Wave w owns query rows (w/2)*16..+15 and key/hd columns (w%2)*32..+31.
// ---------------------------------------------------------------------------
__global__ void attn_kernel(const unsigned short* __restrict__ Q,   // [B,H,T,HD]
                            const unsigned short* __restrict__ Kd,  // [B,H,T,HD]
                            const unsigned short* __restrict__ V,   // [B,H,T,HD]
                            unsigned short* __restrict__ O) {       // [B,T,H*HD]
  const int q0 = blockIdx.x * 64;
  const int h  = blockIdx.y;
  const int b  = blockIdx.z;
  const int t     = threadIdx.x;
  const int wid   = t >> 5;
  const int lane  = t & 31;
  const int lhalf = lane >> 4;
  const int lmod  = lane & 15;

  __shared__ __attribute__((aligned(16))) unsigned short lq[64 * 64];
  __shared__ __attribute__((aligned(16))) unsigned short lk[64 * 64];
  __shared__ __attribute__((aligned(16))) unsigned short lvt[64 * 64]; // [hd][key]
  __shared__ __attribute__((aligned(16))) unsigned short lp[64 * 64];
  __shared__ float ls[64 * 64];
  __shared__ float mrow[64], lrow[64], cf[64];
  __shared__ float pmax[256], psum[256];

  const float slope = exp2f(-0.5f * (float)(h + 1));  // ALiBi, H=16 pow2
  const float scale = 0.125f;                         // 1/sqrt(64)

  const unsigned short* qbase = Q  + (size_t)((b * NH + h) * TSEQ) * HDIM;
  const unsigned short* kbase = Kd + (size_t)((b * NH + h) * TSEQ) * HDIM;
  const unsigned short* vbase = V  + (size_t)((b * NH + h) * TSEQ) * HDIM;

  // Load Q tile (64x64 bf16) via async copies
#pragma unroll
  for (int r = 0; r < 2; ++r) {
    int idx = t + r * 256;
    int row = idx >> 3;
    int cc  = (idx & 7) * 8;
    cp16_g2l(&lq[row * 64 + cc], &qbase[(q0 + row) * HDIM + cc]);
  }
  if (t < 64) { mrow[t] = -1e30f; lrow[t] = 0.f; }
  wait_async0();
  __syncthreads();

  // Preload this wave's Q fragments (2 K-chunks of 32)
  FragB qf[2];
  {
    int m = (wid >> 1) * 16 + lmod;
    const unsigned* p = (const unsigned*)&lq[m * 64];
#pragma unroll
    for (int kc2 = 0; kc2 < 2; ++kc2) {
      int kb = kc2 * 32 + lhalf * 8;
#pragma unroll
      for (int j = 0; j < 4; ++j) {
        qf[kc2].u[j]     = p[(kb >> 1) + j];
        qf[kc2].u[4 + j] = p[((kb + 16) >> 1) + j];
      }
    }
  }

  v8f zf = {};
  v8f o[2]; o[0] = zf; o[1] = zf;

  int kt_start = q0 - WIN; if (kt_start < 0) kt_start = 0;

  for (int kt0 = kt_start; kt0 <= q0; kt0 += 64) {
    __syncthreads();
    // K tile [key][d] via async copies
#pragma unroll
    for (int r = 0; r < 2; ++r) {
      int idx = t + r * 256;
      int row = idx >> 3;
      int cc  = (idx & 7) * 8;
      cp16_g2l(&lk[row * 64 + cc], &kbase[(kt0 + row) * HDIM + cc]);
    }
    // V tile transposed into LDS: lvt[hd][key] (register-staged)
    {
      int key = t >> 2;
      int seg = (t & 3) * 16;
      Stage st;
      st.q[0] = *(const uint4*)(&vbase[(kt0 + key) * HDIM + seg]);
      st.q[1] = *(const uint4*)(&vbase[(kt0 + key) * HDIM + seg + 8]);
#pragma unroll
      for (int j = 0; j < 16; ++j) lvt[(seg + j) * 64 + key] = st.s[j];
    }
    wait_async0();
    __syncthreads();

    // S = Q * K^T  (wave strip 16 rows x 32 key-cols)
    {
      v8f s[2]; s[0] = zf; s[1] = zf;
#pragma unroll
      for (int kc2 = 0; kc2 < 2; ++kc2) {
        FragB kf[2];
#pragma unroll
        for (int ni = 0; ni < 2; ++ni) {
          int n = (wid & 1) * 32 + ni * 16 + lmod;
          const unsigned* p = (const unsigned*)&lk[n * 64 + kc2 * 32 + lhalf * 8];
#pragma unroll
          for (int j = 0; j < 4; ++j) { kf[ni].u[j] = p[j]; kf[ni].u[4 + j] = p[8 + j]; }
        }
#pragma unroll
        for (int ni = 0; ni < 2; ++ni)
          s[ni] = __builtin_amdgcn_wmma_f32_16x16x32_bf16(
              false, qf[kc2].v, false, kf[ni].v, (short)0, s[ni], false, false);
      }
      // scale + ALiBi + causal/window mask -> LDS fp32
#pragma unroll
      for (int ni = 0; ni < 2; ++ni) {
        FragC c; c.v = s[ni];
        int col = (wid & 1) * 32 + ni * 16 + lmod;
#pragma unroll
        for (int e = 0; e < 8; ++e) {
          int row = (wid >> 1) * 16 + e + lhalf * 8;
          int qi = q0 + row, kj = kt0 + col;
          float val = c.f[e] * scale + slope * (float)(kj - qi);
          if (kj > qi || kj < qi - WIN) val = -1e30f;
          ls[row * 64 + col] = val;
        }
      }
    }
    __syncthreads();

    // Online softmax: partial max per (row, 16-col segment)
    {
      int row = t >> 2, seg = t & 3;
      float mx = -1e30f;
#pragma unroll
      for (int j = 0; j < 16; ++j) mx = fmaxf(mx, ls[row * 64 + seg * 16 + j]);
      pmax[t] = mx;
    }
    __syncthreads();
    if (t < 64) {
      float mnew = fmaxf(fmaxf(pmax[t * 4], pmax[t * 4 + 1]),
                         fmaxf(pmax[t * 4 + 2], pmax[t * 4 + 3]));
      mnew = fmaxf(mnew, mrow[t]);
      cf[t] = __expf(mrow[t] - mnew);
      mrow[t] = mnew;
    }
    __syncthreads();
    {
      int row = t >> 2, seg = t & 3;
      float m = mrow[row], sum = 0.f;
#pragma unroll
      for (int j = 0; j < 16; ++j) {
        float e = __expf(ls[row * 64 + seg * 16 + j] - m);
        lp[row * 64 + seg * 16 + j] = f2bf(e);
        sum += e;
      }
      psum[t] = sum;
    }
    __syncthreads();
    if (t < 64)
      lrow[t] = lrow[t] * cf[t] + psum[t * 4] + psum[t * 4 + 1] +
                psum[t * 4 + 2] + psum[t * 4 + 3];

    // Rescale O accumulators by correction factor, then O += P * V
#pragma unroll
    for (int ni = 0; ni < 2; ++ni) {
      FragC c; c.v = o[ni];
#pragma unroll
      for (int e = 0; e < 8; ++e) {
        int row = (wid >> 1) * 16 + e + lhalf * 8;
        c.f[e] *= cf[row];
      }
      o[ni] = c.v;
    }
#pragma unroll
    for (int kc2 = 0; kc2 < 2; ++kc2) {
      FragB pf, vf[2];
      int m = (wid >> 1) * 16 + lmod;
      int kb = kc2 * 32 + lhalf * 8;
      const unsigned* pp = (const unsigned*)&lp[m * 64];
#pragma unroll
      for (int j = 0; j < 4; ++j) {
        pf.u[j]     = pp[(kb >> 1) + j];
        pf.u[4 + j] = pp[((kb + 16) >> 1) + j];
      }
#pragma unroll
      for (int ni = 0; ni < 2; ++ni) {
        int n = (wid & 1) * 32 + ni * 16 + lmod;
        const unsigned* pv = (const unsigned*)&lvt[n * 64 + kc2 * 32 + lhalf * 8];
#pragma unroll
        for (int j = 0; j < 4; ++j) { vf[ni].u[j] = pv[j]; vf[ni].u[4 + j] = pv[8 + j]; }
      }
#pragma unroll
      for (int ni = 0; ni < 2; ++ni)
        o[ni] = __builtin_amdgcn_wmma_f32_16x16x32_bf16(
            false, pf.v, false, vf[ni].v, (short)0, o[ni], false, false);
    }
  }

  __syncthreads();
  // O / l -> bf16 [B,T,H*HD]
#pragma unroll
  for (int ni = 0; ni < 2; ++ni) {
    FragC c; c.v = o[ni];
    int col = (wid & 1) * 32 + ni * 16 + lmod;   // hd index
#pragma unroll
    for (int e = 0; e < 8; ++e) {
      int row = (wid >> 1) * 16 + e + lhalf * 8;
      float val = c.f[e] / lrow[row];
      O[(size_t)(b * TSEQ + q0 + row) * TDIM + h * HDIM + col] = f2bf(val);
    }
  }
}

// ---------------------------------------------------------------------------
// Host-side orchestration
// ---------------------------------------------------------------------------
extern "C" void kernel_launch(void* const* d_in, const int* in_sizes, int n_in,
                              void* d_out, int out_size, void* d_ws, size_t ws_size,
                              hipStream_t stream) {
  (void)in_sizes; (void)n_in; (void)out_size; (void)ws_size;
  const float* x   = (const float*)d_in[0];
  const float* wq  = (const float*)d_in[1];
  const float* wk  = (const float*)d_in[2];
  const float* wv  = (const float*)d_in[3];
  const float* wo  = (const float*)d_in[4];
  const float* qnw = (const float*)d_in[5];
  const float* knw = (const float*)d_in[6];
  float* out = (float*)d_out;

  char* ws = (char*)d_ws;
  const size_t MB = 1u << 20;
  unsigned short* xbf  = (unsigned short*)(ws + 0);          //  8 MB
  unsigned short* wqbf = (unsigned short*)(ws + 8  * MB);    //  2 MB
  unsigned short* wkbf = (unsigned short*)(ws + 10 * MB);
  unsigned short* wvbf = (unsigned short*)(ws + 12 * MB);
  unsigned short* wobf = (unsigned short*)(ws + 14 * MB);
  float*          q32  = (float*)(ws + 16 * MB);             // 16 MB
  float*          k32  = (float*)(ws + 32 * MB);             // 16 MB
  unsigned short* qbf  = (unsigned short*)(ws + 48 * MB);    //  8 MB [B,H,T,HD]
  unsigned short* kbf  = (unsigned short*)(ws + 56 * MB);
  unsigned short* vbf  = (unsigned short*)(ws + 64 * MB);
  unsigned short* abf  = (unsigned short*)(ws + 16 * MB);    // alias q32 (dead)

  const int M = NB * TSEQ;        // 4096
  const int nx = M * TDIM;        // 4 M elems
  const int nw = TDIM * TDIM;     // 1 M elems

  cvt_f32_bf16<<<nx / 1024, 256, 0, stream>>>(x,  xbf,  nx);
  cvt_f32_bf16<<<nw / 1024, 256, 0, stream>>>(wq, wqbf, nw);
  cvt_f32_bf16<<<nw / 1024, 256, 0, stream>>>(wk, wkbf, nw);
  cvt_f32_bf16<<<nw / 1024, 256, 0, stream>>>(wv, wvbf, nw);
  cvt_f32_bf16<<<nw / 1024, 256, 0, stream>>>(wo, wobf, nw);

  dim3 gg(M / 128, TDIM / 64);
  gemm_bf16<<<gg, 256, 0, stream>>>(xbf, wqbf, q32, nullptr, M, TDIM, TDIM, 0);
  gemm_bf16<<<gg, 256, 0, stream>>>(xbf, wkbf, k32, nullptr, M, TDIM, TDIM, 0);
  gemm_bf16<<<gg, 256, 0, stream>>>(xbf, wvbf, nullptr, vbf, M, TDIM, TDIM, 1);

  rmsnorm_pack<<<M, 256, 0, stream>>>(q32, qnw, qbf);
  rmsnorm_pack<<<M, 256, 0, stream>>>(k32, knw, kbf);

  dim3 ag(TSEQ / 64, NH, NB);
  attn_kernel<<<ag, 256, 0, stream>>>(qbf, kbf, vbf, abf);

  gemm_bf16<<<gg, 256, 0, stream>>>(abf, wobf, out, nullptr, M, TDIM, TDIM, 0);
}
